// AttentionBlock_32950989095003
// MI455X (gfx1250) — compile-verified, matrix operand order checked by reading, not run on hardware
//
#include <hip/hip_runtime.h>

// ---------------------------------------------------------------------------
// Problem constants (from reference): B=4, T=4096, D=1024, H=16, g=4
// ---------------------------------------------------------------------------
#define B_    4
#define T_    4096
#define D_    1024
#define H_    16
#define G_    4
#define DH_   256     // head dim = g*D/H
#define TG_   1024    // grouped timesteps = T/g
#define LREL_ 2047    // 2*Tg - 1
#define PER_  8188    // rows of pe matrix

typedef __bf16 bf16_t;
typedef bf16_t v16bf __attribute__((ext_vector_type(16)));
typedef bf16_t v8bf  __attribute__((ext_vector_type(8)));
typedef float  v8f   __attribute__((ext_vector_type(8)));

// float -> bf16 bits (round-to-nearest-even)
__device__ __forceinline__ unsigned short f2bf(float f) {
  union { float f; unsigned int u; } c; c.f = f;
  unsigned int u = c.u;
  return (unsigned short)((u + 0x7FFFu + ((u >> 16) & 1u)) >> 16);
}

// Load one WMMA bf16 operand (A or B^T, both row-major along K).
// lane: row = lane%16 (caller bakes into rowptr), khalf = lane/16.
__device__ __forceinline__ v16bf ld16(const unsigned short* rowptr, int k0, int khalf) {
  v8bf lo = *reinterpret_cast<const v8bf*>(rowptr + k0 + khalf * 8);
  v8bf hi = *reinterpret_cast<const v8bf*>(rowptr + k0 + 16 + khalf * 8);
  v16bf r;
#pragma unroll
  for (int i = 0; i < 8; ++i) { r[i] = lo[i]; r[i + 8] = hi[i]; }
  return r;
}

__device__ __forceinline__ v8f vzero8() {
  return (v8f){0.f, 0.f, 0.f, 0.f, 0.f, 0.f, 0.f, 0.f};
}

// 16x16 GEMM over K=256 with pre-loaded A operand; B double-buffered so each
// WMMA only depends on the load issued one step earlier (partial loadcnt waits).
__device__ __forceinline__ v8f gemm_pre(const v16bf* a8, const unsigned short* brow,
                                        int khalf) {
  v8f c = vzero8();
  v16bf b = ld16(brow, 0, khalf);
#pragma unroll
  for (int k = 0; k < 8; ++k) {
    int kn = (k < 7) ? (k + 1) * 32 : 0;      // tail reload is dead
    v16bf bn = ld16(brow, kn, khalf);
    c = __builtin_amdgcn_wmma_f32_16x16x32_bf16(false, a8[k], false, b,
                                                (short)0, c, false, false);
    b = bn;
  }
  return c;
}

// ---------------------------------------------------------------------------
// 1) weight transpose+convert: Wt_bf16[n][k] = W_f32[k][n]   (D x D)
// ---------------------------------------------------------------------------
__global__ void k_transpose(const float* __restrict__ W, unsigned short* __restrict__ Wt) {
  size_t idx = (size_t)blockIdx.x * 256 + threadIdx.x;
  if (idx >= (size_t)D_ * D_) return;
  int n = (int)(idx >> 10), k = (int)(idx & 1023);
  Wt[idx] = f2bf(W[(size_t)k * D_ + n]);
}

// ---------------------------------------------------------------------------
// 2) LayerNorm rows of xs (B*T rows, D cols) -> bf16 xn
// ---------------------------------------------------------------------------
__global__ void k_layernorm(const float* __restrict__ xs, const float* __restrict__ g,
                            const float* __restrict__ bta, unsigned short* __restrict__ xn) {
  __shared__ float red[256];
  int row = blockIdx.x, t = threadIdx.x;
  const float* x = xs + (size_t)row * D_;
  float v[4], s = 0.f;
#pragma unroll
  for (int i = 0; i < 4; ++i) { v[i] = x[t + i * 256]; s += v[i]; }
  red[t] = s; __syncthreads();
  for (int o = 128; o > 0; o >>= 1) { if (t < o) red[t] += red[t + o]; __syncthreads(); }
  float mu = red[0] * (1.f / D_); __syncthreads();
  float s2 = 0.f;
#pragma unroll
  for (int i = 0; i < 4; ++i) { float d = v[i] - mu; s2 += d * d; }
  red[t] = s2; __syncthreads();
  for (int o = 128; o > 0; o >>= 1) { if (t < o) red[t] += red[t + o]; __syncthreads(); }
  float inv = rsqrtf(red[0] * (1.f / D_) + 1e-5f);
#pragma unroll
  for (int i = 0; i < 4; ++i) {
    int c = t + i * 256;
    xn[(size_t)row * D_ + c] = f2bf((v[i] - mu) * inv * g[c] + bta[c]);
  }
}

// ---------------------------------------------------------------------------
// 3) relative positional encoding matrix pe (PER_ x D) in bf16
// ---------------------------------------------------------------------------
__global__ void k_pe(unsigned short* __restrict__ pe) {
  size_t idx = (size_t)blockIdx.x * 256 + threadIdx.x;
  if (idx >= (size_t)PER_ * D_) return;
  int p = (int)(idx >> 10), d = (int)(idx & 1023);
  float pos = (float)(T_ - 1 - p);
  pos = fminf(fmaxf(pos, -10000.f), 10000.f);
  float freq = __expf(-((float)(d & 1022)) * (9.210340371976184f / (float)D_));
  float ang = pos * freq;
  pe[idx] = f2bf((d & 1) ? __cosf(ang) : __sinf(ang));
}

// ---------------------------------------------------------------------------
// Register-blocked 16(M) x 64(N) GEMM body over K. A loaded once per k-step
// and reused by 4 WMMAs (1.25 operand loads / WMMA); all operands software
// double-buffered: step i's WMMAs consume registers loaded at step i-1 while
// step i's 10 loads are in flight.
// ---------------------------------------------------------------------------
__device__ __forceinline__ void gemm_16x64(const unsigned short* arow,
                                           const unsigned short* b0p,
                                           const unsigned short* b1p,
                                           const unsigned short* b2p,
                                           const unsigned short* b3p,
                                           int K, int khalf, v8f c[4]) {
#pragma unroll
  for (int nt = 0; nt < 4; ++nt) c[nt] = vzero8();
  v16bf a  = ld16(arow, 0, khalf);
  v16bf b0 = ld16(b0p, 0, khalf);
  v16bf b1 = ld16(b1p, 0, khalf);
  v16bf b2 = ld16(b2p, 0, khalf);
  v16bf b3 = ld16(b3p, 0, khalf);
#pragma unroll 4
  for (int k0 = 0; k0 < K; k0 += 32) {
    int kn = (k0 + 32 < K) ? (k0 + 32) : 0;   // tail reload is dead
    __builtin_prefetch(arow + k0 + 512, 0, 0);   // global_prefetch_b8
    v16bf an  = ld16(arow, kn, khalf);
    v16bf nb0 = ld16(b0p, kn, khalf);
    v16bf nb1 = ld16(b1p, kn, khalf);
    v16bf nb2 = ld16(b2p, kn, khalf);
    v16bf nb3 = ld16(b3p, kn, khalf);
    c[0] = __builtin_amdgcn_wmma_f32_16x16x32_bf16(false, a, false, b0, (short)0, c[0], false, false);
    c[1] = __builtin_amdgcn_wmma_f32_16x16x32_bf16(false, a, false, b1, (short)0, c[1], false, false);
    c[2] = __builtin_amdgcn_wmma_f32_16x16x32_bf16(false, a, false, b2, (short)0, c[2], false, false);
    c[3] = __builtin_amdgcn_wmma_f32_16x16x32_bf16(false, a, false, b3, (short)0, c[3], false, false);
    a = an; b0 = nb0; b1 = nb1; b2 = nb2; b3 = nb3;
  }
}

// ---------------------------------------------------------------------------
// 4) QKV GEMM: (B*T x D) @ Wt^T + bias, epilogue reshapes to head layout.
//    mode 0: K heads row-major [bh][tg][e]
//    mode 1: V transposed      [bh][e][tg]
//    mode 2: Q -> qu=bf16(q+u), qv=bf16(q+v), both [bh][tg][e]
//    wave tile = 16M x 64N; block = 8 waves = 16M x 512N
// ---------------------------------------------------------------------------
__global__ void k_gemm_qkv(const unsigned short* __restrict__ A,
                           const unsigned short* __restrict__ Bt,
                           const float* __restrict__ bias, int mode,
                           const float* __restrict__ uvec, const float* __restrict__ vvec,
                           unsigned short* __restrict__ out0,
                           unsigned short* __restrict__ out1) {
  int lane = threadIdx.x & 31, wid = threadIdx.x >> 5;
  int col = lane & 15, khalf = lane >> 4;
  int m0 = blockIdx.y * 16;
  int n0 = (blockIdx.x * 8 + wid) * 64;
  const unsigned short* arow = A + (size_t)(m0 + col) * D_;
  v8f c[4];
  gemm_16x64(arow,
             Bt + (size_t)(n0 +  0 + col) * D_, Bt + (size_t)(n0 + 16 + col) * D_,
             Bt + (size_t)(n0 + 32 + col) * D_, Bt + (size_t)(n0 + 48 + col) * D_,
             D_, khalf, c);
#pragma unroll
  for (int nt = 0; nt < 4; ++nt) {
    int j = n0 + nt * 16 + col;
    float bj = bias[j];
#pragma unroll
    for (int r = 0; r < 8; ++r) {
      int mrow = m0 + khalf * 8 + r;
      float val = c[nt][r] + bj;
      int b  = mrow >> 12;          // / T_
      int tb = mrow & (T_ - 1);
      int tg = tb >> 2;
      int h  = (tb & 3) * 4 + (j >> 8);
      int e  = j & 255;
      size_t bh = (size_t)b * H_ + h;
      if (mode == 0) {
        out0[(bh * TG_ + tg) * DH_ + e] = f2bf(val);
      } else if (mode == 1) {
        out0[(bh * DH_ + e) * TG_ + tg] = f2bf(val);
      } else {
        out0[(bh * TG_ + tg) * DH_ + e] = f2bf(val + uvec[h * DH_ + e]);
        out1[(bh * TG_ + tg) * DH_ + e] = f2bf(val + vvec[h * DH_ + e]);
      }
    }
  }
}

// ---------------------------------------------------------------------------
// 5) E GEMM: pe (PER_ x D) @ Wpos^T, epilogue reshape -> E[h][l][e] bf16
// ---------------------------------------------------------------------------
__global__ void k_gemm_e(const unsigned short* __restrict__ pe,
                         const unsigned short* __restrict__ Bt,
                         unsigned short* __restrict__ Emat) {
  int lane = threadIdx.x & 31, wid = threadIdx.x >> 5;
  int col = lane & 15, khalf = lane >> 4;
  int m0 = blockIdx.y * 16;
  int n0 = (blockIdx.x * 8 + wid) * 64;
  int arow = m0 + col; if (arow > PER_ - 1) arow = PER_ - 1;
  v8f c[4];
  gemm_16x64(pe + (size_t)arow * D_,
             Bt + (size_t)(n0 +  0 + col) * D_, Bt + (size_t)(n0 + 16 + col) * D_,
             Bt + (size_t)(n0 + 32 + col) * D_, Bt + (size_t)(n0 + 48 + col) * D_,
             D_, khalf, c);
#pragma unroll
  for (int nt = 0; nt < 4; ++nt) {
    int j = n0 + nt * 16 + col;
#pragma unroll
    for (int r = 0; r < 8; ++r) {
      int p = m0 + khalf * 8 + r;
      if (p < PER_) {
        int l = p >> 2;
        int h = (p & 3) * 4 + (j >> 8);
        int e = j & 255;
        Emat[((size_t)h * LREL_ + l) * DH_ + e] = f2bf(c[nt][r]);
      }
    }
  }
}

// ---------------------------------------------------------------------------
// 6) Flash attention with rel-shift. 128 threads = 4 waves, each wave owns a
//    16-row query tile. Q operands ((q+u), (q+v)) are hoisted into registers
//    once; per 32-key step: QK^T (16 wmma) + Q.E^T band (32 wmma) + online
//    softmax + P@V (16 wmma over dh=256, V streamed double-buffered).
// ---------------------------------------------------------------------------
__global__ void k_attn(const unsigned short* __restrict__ qu,
                       const unsigned short* __restrict__ qv,
                       const unsigned short* __restrict__ kh,
                       const unsigned short* __restrict__ vT,
                       const unsigned short* __restrict__ Emat,
                       const unsigned char* __restrict__ masks,
                       unsigned short* __restrict__ attnout) {
  __shared__ __align__(16) float          xbuf[4][2][16][16]; // X1/X2 staging
  __shared__ __align__(16) unsigned short pbuf[4][16][32];    // P staging (A layout src)
  int lane = threadIdx.x & 31, wid = threadIdx.x >> 5;
  int col = lane & 15, khalf = lane >> 4;
  int bh   = blockIdx.x >> 4;          // 0..63
  int qblk = blockIdx.x & 15;
  int b = bh >> 4, h = bh & 15;
  int q0 = (qblk * 4 + wid) * 16;

  const unsigned short* quRow = qu + ((size_t)bh * TG_ + q0 + col) * DH_;
  const unsigned short* qvRow = qv + ((size_t)bh * TG_ + q0 + col) * DH_;
  const unsigned short* kB = kh + (size_t)bh * TG_ * DH_;
  const unsigned short* vB = vT + (size_t)bh * DH_ * TG_;
  const unsigned short* EB = Emat + (size_t)h * LREL_ * DH_;

  // hoist Q operands (A layout) for the whole key loop
  v16bf qa[8], va[8];
#pragma unroll
  for (int k = 0; k < 8; ++k) {
    qa[k] = ld16(quRow, k * 32, khalf);
    va[k] = ld16(qvRow, k * 32, khalf);
  }

  v8f acc[16];
#pragma unroll
  for (int nt = 0; nt < 16; ++nt) acc[nt] = vzero8();
  float mrun[8], lrun[8], sreg[2][8];
#pragma unroll
  for (int r = 0; r < 8; ++r) { mrun[r] = -1e30f; lrun[r] = 0.f; }

  for (int kb = 0; kb < 32; ++kb) {
#pragma unroll
    for (int sub = 0; sub < 2; ++sub) {
      int n0 = (kb * 2 + sub) * 16;
      // content scores: (q+u) . k^T
      v8f ck = gemm_pre(qa, kB + (size_t)(n0 + col) * DH_, khalf);
      // positional band: X2 = (q+v).E[rbase..rbase+16)^T, X1 = ...[rbase-16..rbase)
      int rbase = (TG_ - 1) + n0 - q0;
      int r2 = rbase + col;        if (r2 > LREL_ - 1) r2 = LREL_ - 1;
      int r1 = rbase - 16 + col;   if (r1 < 0) r1 = 0;
      v8f cx2 = gemm_pre(va, EB + (size_t)r2 * DH_, khalf);
      v8f cx1 = gemm_pre(va, EB + (size_t)r1 * DH_, khalf);
      __syncthreads();
#pragma unroll
      for (int r = 0; r < 8; ++r) {
        xbuf[wid][1][khalf * 8 + r][col] = cx2[r];
        xbuf[wid][0][khalf * 8 + r][col] = cx1[r];
      }
      __syncthreads();
      float maskadd = masks[(size_t)b * T_ + (size_t)(n0 + col) * G_] ? 0.f : -1e9f;
#pragma unroll
      for (int r = 0; r < 8; ++r) {
        int dq = khalf * 8 + r;
        int dd = col - dq;         // rel_shift gather: l = rbase + (k - q)
        float se = (dd >= 0) ? xbuf[wid][1][dq][dd] : xbuf[wid][0][dq][16 + dd];
        sreg[sub][r] = (ck[r] + se) * 0.0625f + maskadd;   // /sqrt(256)
      }
    }
    // online softmax over this 32-key chunk (row = 16 lanes within half)
    float alpha[8];
#pragma unroll
    for (int r = 0; r < 8; ++r) {
      float mloc = fmaxf(sreg[0][r], sreg[1][r]);
      for (int m = 1; m < 16; m <<= 1) mloc = fmaxf(mloc, __shfl_xor(mloc, m, 32));
      float mnew = fmaxf(mrun[r], mloc);
      float a0 = __expf(mrun[r] - mnew);
      float p0 = __expf(sreg[0][r] - mnew);
      float p1 = __expf(sreg[1][r] - mnew);
      float ss = p0 + p1;
      for (int m = 1; m < 16; m <<= 1) ss += __shfl_xor(ss, m, 32);
      lrun[r] = lrun[r] * a0 + ss;
      mrun[r] = mnew; alpha[r] = a0;
      sreg[0][r] = p0; sreg[1][r] = p1;
    }
    // stage P (16x32) to LDS, reload as WMMA A operand
    __syncthreads();
#pragma unroll
    for (int r = 0; r < 8; ++r) {
      pbuf[wid][khalf * 8 + r][col]      = f2bf(sreg[0][r]);
      pbuf[wid][khalf * 8 + r][16 + col] = f2bf(sreg[1][r]);
    }
    __syncthreads();
    v16bf pa = ld16(&pbuf[wid][col][0], 0, khalf);
    // rescale accumulator, then P @ V over dh (V stored transposed [e][tg]);
    // V operand double-buffered across the 16 N-tiles
    const unsigned short* vbase = vB + (size_t)col * TG_ + kb * 32;
    v16bf vb = ld16(vbase, 0, khalf);
#pragma unroll
    for (int nt = 0; nt < 16; ++nt) {
      int ntn = (nt < 15) ? (nt + 1) : 0;      // tail reload is dead
      v16bf vbn = ld16(vbase + (size_t)ntn * 16 * TG_, 0, khalf);
      v8f c = acc[nt];
#pragma unroll
      for (int r = 0; r < 8; ++r) c[r] *= alpha[r];
      acc[nt] = __builtin_amdgcn_wmma_f32_16x16x32_bf16(false, pa, false, vb,
                                                        (short)0, c, false, false);
      vb = vbn;
    }
  }
  // finalize: /l, undo head reshape, store bf16 (B,T,D)
#pragma unroll
  for (int nt = 0; nt < 16; ++nt) {
#pragma unroll
    for (int r = 0; r < 8; ++r) {
      float val = acc[nt][r] / lrun[r];
      int tg = q0 + khalf * 8 + r;
      int e  = nt * 16 + col;
      int rem = h * DH_ + e;
      int t = tg * G_ + (rem >> 10);
      int j = rem & 1023;
      attnout[((size_t)b * T_ + t) * D_ + j] = f2bf(val);
    }
  }
}

// ---------------------------------------------------------------------------
// 7) output GEMM: attn (B*T x D) @ Wo^T + bo -> f32 d_out (16M x 64N / wave)
// ---------------------------------------------------------------------------
__global__ void k_gemm_out(const unsigned short* __restrict__ A,
                           const unsigned short* __restrict__ Bt,
                           const float* __restrict__ bo, float* __restrict__ out) {
  int lane = threadIdx.x & 31, wid = threadIdx.x >> 5;
  int col = lane & 15, khalf = lane >> 4;
  int m0 = blockIdx.y * 16;
  int n0 = (blockIdx.x * 8 + wid) * 64;
  v8f c[4];
  gemm_16x64(A + (size_t)(m0 + col) * D_,
             Bt + (size_t)(n0 +  0 + col) * D_, Bt + (size_t)(n0 + 16 + col) * D_,
             Bt + (size_t)(n0 + 32 + col) * D_, Bt + (size_t)(n0 + 48 + col) * D_,
             D_, khalf, c);
#pragma unroll
  for (int nt = 0; nt < 4; ++nt) {
    int j = n0 + nt * 16 + col;
    float bj = bo[j];
#pragma unroll
    for (int r = 0; r < 8; ++r)
      out[(size_t)(m0 + khalf * 8 + r) * D_ + j] = c[nt][r] + bj;
  }
}

// ---------------------------------------------------------------------------
extern "C" void kernel_launch(void* const* d_in, const int* in_sizes, int n_in,
                              void* d_out, int out_size, void* d_ws, size_t ws_size,
                              hipStream_t stream) {
  const float* xs   = (const float*)d_in[0];
  const unsigned char* masks = (const unsigned char*)d_in[1];   // jnp.bool_ = 1 byte
  const float* lnS  = (const float*)d_in[2];
  const float* lnB  = (const float*)d_in[3];
  const float* Wq   = (const float*)d_in[4];
  const float* bq   = (const float*)d_in[5];
  const float* Wk   = (const float*)d_in[6];
  const float* bk   = (const float*)d_in[7];
  const float* Wv   = (const float*)d_in[8];
  const float* bv   = (const float*)d_in[9];
  const float* Wpos = (const float*)d_in[10];
  const float* u    = (const float*)d_in[11];
  const float* v    = (const float*)d_in[12];
  const float* Wo   = (const float*)d_in[13];
  const float* bo   = (const float*)d_in[14];
  float* out = (float*)d_out;

  char* ws = (char*)d_ws;
  size_t off = 0;
  auto take = [&](size_t bytes) -> void* {
    void* p = ws + off; off += (bytes + 255) & ~(size_t)255; return p;
  };
  unsigned short* xn    = (unsigned short*)take((size_t)B_ * T_ * D_ * 2);
  unsigned short* WqT   = (unsigned short*)take((size_t)D_ * D_ * 2);
  unsigned short* WkT   = (unsigned short*)take((size_t)D_ * D_ * 2);
  unsigned short* WvT   = (unsigned short*)take((size_t)D_ * D_ * 2);
  unsigned short* WposT = (unsigned short*)take((size_t)D_ * D_ * 2);
  unsigned short* WoT   = (unsigned short*)take((size_t)D_ * D_ * 2);
  unsigned short* pe    = (unsigned short*)take((size_t)PER_ * D_ * 2);
  unsigned short* quA   = (unsigned short*)take((size_t)B_ * H_ * TG_ * DH_ * 2);
  unsigned short* qvA   = (unsigned short*)take((size_t)B_ * H_ * TG_ * DH_ * 2);
  unsigned short* kA    = (unsigned short*)take((size_t)B_ * H_ * TG_ * DH_ * 2);
  unsigned short* vTA   = (unsigned short*)take((size_t)B_ * H_ * DH_ * TG_ * 2);
  unsigned short* Emat  = (unsigned short*)take((size_t)H_ * LREL_ * DH_ * 2);
  unsigned short* attn  = (unsigned short*)take((size_t)B_ * T_ * D_ * 2);

  // weights -> bf16 transposed
  int tgrid = (D_ * D_ + 255) / 256;
  k_transpose<<<tgrid, 256, 0, stream>>>(Wq, WqT);
  k_transpose<<<tgrid, 256, 0, stream>>>(Wk, WkT);
  k_transpose<<<tgrid, 256, 0, stream>>>(Wv, WvT);
  k_transpose<<<tgrid, 256, 0, stream>>>(Wpos, WposT);
  k_transpose<<<tgrid, 256, 0, stream>>>(Wo, WoT);

  // layernorm -> bf16 xn
  k_layernorm<<<B_ * T_, 256, 0, stream>>>(xs, lnS, lnB, xn);

  // positional encodings
  k_pe<<<((size_t)PER_ * D_ + 255) / 256, 256, 0, stream>>>(pe);

  // QKV projections with fused head reshape (+u/+v for Q, transpose for V)
  dim3 g1(D_ / 512, (B_ * T_) / 16);
  k_gemm_qkv<<<g1, 256, 0, stream>>>(xn, WqT, bq, 2, u, v, quA, qvA);
  k_gemm_qkv<<<g1, 256, 0, stream>>>(xn, WkT, bk, 0, nullptr, nullptr, kA, nullptr);
  k_gemm_qkv<<<g1, 256, 0, stream>>>(xn, WvT, bv, 1, nullptr, nullptr, vTA, nullptr);

  // E = pe @ Wpos, reshaped to [h][l][e]
  dim3 g2(D_ / 512, (PER_ + 15) / 16);
  k_gemm_e<<<g2, 256, 0, stream>>>(pe, WposT, Emat);

  // flash attention with rel-shift
  k_attn<<<B_ * H_ * (TG_ / 64), 128, 0, stream>>>(quA, qvA, kA, vTA, Emat, masks, attn);

  // output projection
  dim3 g3(D_ / 512, (B_ * T_) / 16);
  k_gemm_out<<<g3, 256, 0, stream>>>(attn, WoT, bo, out);
}
